// ImprovedValueHead_60318520705093
// MI455X (gfx1250) — matrix operand order repaired and checked by reading, not compiled, not run on hardware
//
#include <hip/hip_runtime.h>
#include <math.h>

typedef __attribute__((ext_vector_type(16))) _Float16 v16h;
typedef __attribute__((ext_vector_type(8)))  float    v8f;

#define Hdim 256
#define Ghm  256
#define Nn   16384
#define Ee   262144
#define ROWS (Nn + Ee)   // 278528

// ---- order-preserving float<->uint codes (monotonic; 0 is the minimum) ----
__device__ __forceinline__ unsigned int fcode(float f) {
    unsigned int u = __float_as_uint(f);
    return (u & 0x80000000u) ? ~u : (u | 0x80000000u);
}
__device__ __forceinline__ float fdecode(unsigned int c) {
    return (c & 0x80000000u) ? __uint_as_float(c ^ 0x80000000u)
                             : __uint_as_float(~c);
}
__device__ __forceinline__ float gelu_exact(float x) {
    return 0.5f * x * (1.0f + erff(x * 0.70710678118654752440f));
}

// ============================================================================
// K0: convert aw1 (f32 [256][128]) into f16, pre-swizzled into WMMA B-fragment
// order: frag(nt,kc) -> 32 lanes x 16 halves.  Contiguous-k B order used here (A matches):
//   lane = (n%16) + 16*(kin/16),  half = kin%16   (kin = k within 32-chunk)
// Also zero the per-graph max codes.
// ============================================================================
__global__ void prep_kernel(const float* __restrict__ aw1,
                            _Float16* __restrict__ aw1h,
                            unsigned int* __restrict__ mcode) {
    const int t = threadIdx.x;
    if (t < Ghm) mcode[t] = 0u;
    for (int i = 0; i < 128; ++i) {
        int idx = t * 128 + i;         // 256*128 = 32768 elements
        int k = idx >> 7;              // 0..255
        int n = idx & 127;             // 0..127
        int kc  = k >> 5;              // K-chunk 0..7
        int kin = k & 31;
        int nt  = n >> 4;              // N-tile 0..7
        int lane = (n & 15) + ((kin >> 4) << 4);
        int half = kin & 15;
        aw1h[((((nt << 3) + kc) << 5) + lane) * 16 + half] = (_Float16)aw1[idx];
    }
}

// ============================================================================
// K1: attention scores via v_wmma_f32_16x16x32_f16.
// One wave per 16-row tile of feats=[node|edge]; 8 N-tiles x 8 K-chunks.
// Epilogue: +ab1, tanh, dot with aw2, lane-butterfly reduce, +ab2,
// store score & segment id, atomicMax per-graph max code.
// ============================================================================
__global__ void __launch_bounds__(256)
attn_kernel(const float* __restrict__ nodef, const float* __restrict__ edgef,
            const int* __restrict__ eidx, const int* __restrict__ batch,
            const _Float16* __restrict__ aw1h,
            const float* __restrict__ ab1, const float* __restrict__ aw2,
            const float* __restrict__ ab2,
            float* __restrict__ scores, int* __restrict__ segarr,
            unsigned int* __restrict__ mcode) {
    __shared__ _Float16 sB[32768];   // 64 KB: all 64 B-fragments

    // stage pre-swizzled B into LDS (coalesced 16B copies)
    {
        const float4* s4 = (const float4*)aw1h;
        float4* d4 = (float4*)sB;
        for (int i = threadIdx.x; i < 4096; i += 256) d4[i] = s4[i];
    }
    __syncthreads();

    const int wave = threadIdx.x >> 5;
    const int lane = threadIdx.x & 31;
    const int tile = blockIdx.x * 8 + wave;
    const long rowbase = (long)tile * 16;
    const int mrow = lane & 15;
    const int gsel = lane >> 4;

    // ---- build 8 A-fragments (16-bit A 16x32 layout) from global f32 ----
    const long r = rowbase + mrow;
    const float* rp = (r < Nn) ? (nodef + r * Hdim)
                               : (edgef + (r - (long)Nn) * Hdim);
    v16h af[8];
#pragma unroll
    for (int kc = 0; kc < 8; ++kc) {
        const int k0 = kc * 32 + gsel * 16;
        float4 a0 = *(const float4*)(rp + k0);
        float4 a1 = *(const float4*)(rp + k0 + 4);
        float4 b0 = *(const float4*)(rp + k0 + 8);
        float4 b1 = *(const float4*)(rp + k0 + 12);
        v16h a;
        a[0]=(_Float16)a0.x; a[1]=(_Float16)a0.y; a[2]=(_Float16)a0.z; a[3]=(_Float16)a0.w;
        a[4]=(_Float16)a1.x; a[5]=(_Float16)a1.y; a[6]=(_Float16)a1.z; a[7]=(_Float16)a1.w;
        a[8]=(_Float16)b0.x; a[9]=(_Float16)b0.y; a[10]=(_Float16)b0.z; a[11]=(_Float16)b0.w;
        a[12]=(_Float16)b1.x; a[13]=(_Float16)b1.y; a[14]=(_Float16)b1.z; a[15]=(_Float16)b1.w;
        af[kc] = a;
    }

    float sacc[8] = {0.f,0.f,0.f,0.f,0.f,0.f,0.f,0.f};

    for (int nt = 0; nt < 8; ++nt) {
        v8f acc = {0.f,0.f,0.f,0.f,0.f,0.f,0.f,0.f};
#pragma unroll
        for (int kc = 0; kc < 8; ++kc) {
            const float4* bp =
                (const float4*)(sB + ((((nt << 3) + kc) << 5) + lane) * 16);
            union { float4 f[2]; v16h h; } u;
            u.f[0] = bp[0]; u.f[1] = bp[1];
            acc = __builtin_amdgcn_wmma_f32_16x16x32_f16(
                false, af[kc], false, u.h, (short)0, acc, false, false);
        }
        const int col = nt * 16 + mrow;            // D: N = lane%16
        const float bb = ab1[col];
        const float a2 = aw2[col];
#pragma unroll
        for (int v = 0; v < 8; ++v)
            sacc[v] += tanhf(acc[v] + bb) * a2;    // D: M = v + 8*gsel
    }

    // reduce over the 16 columns (lane bits 0..3)
#pragma unroll
    for (int v = 0; v < 8; ++v) {
        float s = sacc[v];
        s += __shfl_xor(s, 1, 32);
        s += __shfl_xor(s, 2, 32);
        s += __shfl_xor(s, 4, 32);
        s += __shfl_xor(s, 8, 32);
        sacc[v] = s;
    }

    if (mrow == 0) {                // lanes 0 (rows 0..7) and 16 (rows 8..15)
        const float b2 = ab2[0];
        for (int v = 0; v < 8; ++v) {
            const long rr = rowbase + gsel * 8 + v;
            const float s = sacc[v] + b2;
            scores[rr] = s;
            int sg;
            if (rr < Nn) sg = batch[rr];
            else         sg = batch[eidx[rr - Nn]];   // edge_index[0][e]
            segarr[rr] = sg;
            atomicMax(&mcode[sg], fcode(s));
        }
    }
}

// ============================================================================
// K2: per-graph softmax denom + weighted segment-sum pooling (deterministic).
// Block g: nodes are rows [64g,64g+64); edges found by scanning segarr
// (1 MB, L2-resident) with ballot-compaction per 256-edge chunk.
// Thread t owns pooled column t.
// ============================================================================
__global__ void __launch_bounds__(256)
pool_kernel(const float* __restrict__ nodef, const float* __restrict__ edgef,
            const float* __restrict__ scores, const int* __restrict__ segarr,
            const unsigned int* __restrict__ mcode, float* __restrict__ pooled) {
    const int g = blockIdx.x;
    const int t = threadIdx.x;
    const int lane = t & 31, wave = t >> 5;

    __shared__ float sred[256];
    __shared__ float swnode[64];
    __shared__ int   scnt[8];
    __shared__ int   slist[256];
    __shared__ float sw[256];
    __shared__ float sdenom;

    const float m = fdecode(mcode[g]);

    // ---- denom: fixed-order tree reduction -> deterministic ----
    float partial = 0.f;
    if (t < 64) partial += expf(scores[g * 64 + t] - m);
    for (int i = t; i < Ee; i += 256)
        if (segarr[Nn + i] == g) partial += expf(scores[Nn + i] - m);
    sred[t] = partial;
    __syncthreads();
    for (int s = 128; s > 0; s >>= 1) {
        if (t < s) sred[t] += sred[t + s];
        __syncthreads();
    }
    if (t == 0) sdenom = sred[0];
    __syncthreads();
    const float denom = sdenom;
    if (t < 64) swnode[t] = expf(scores[g * 64 + t] - m) / denom;
    __syncthreads();

    // ---- nodes: 64 contiguous rows, coalesced ----
    float acc = 0.f;
    const float* np = nodef + (size_t)g * 64 * Hdim;
    for (int i = 0; i < 64; ++i) acc += swnode[i] * np[i * Hdim + t];

    // ---- edges: chunked ballot compaction (deterministic order) ----
    const unsigned int lmask = (1u << lane) - 1u;
    for (int c = 0; c < Ee / 256; ++c) {
        const int eid = c * 256 + t;
        const bool match = (segarr[Nn + eid] == g);
        const unsigned int wm = (unsigned int)__ballot(match);
        if (lane == 0) scnt[wave] = __popc(wm);
        __syncthreads();
        int base = 0, total = 0;
#pragma unroll
        for (int w = 0; w < 8; ++w) {
            total += scnt[w];
            if (w < wave) base += scnt[w];
        }
        if (match) {
            const int pos = base + __popc(wm & lmask);
            slist[pos] = eid;
            sw[pos] = expf(scores[Nn + eid] - m) / denom;
        }
        __syncthreads();
        for (int j = 0; j < total; ++j)
            acc += sw[j] * edgef[(size_t)slist[j] * Hdim + t];
        __syncthreads();
    }

    pooled[(size_t)g * Hdim + t] = acc;
}

// ============================================================================
// K3: value head. Block per graph, thread per hidden column.
// LayerNorm -> (256x256) GEMM -> GELU -> (256x128) -> GELU -> dot(128) -> tanh
// ============================================================================
__global__ void __launch_bounds__(256)
head_kernel(const float* __restrict__ pooled,
            const float* __restrict__ ln_g, const float* __restrict__ ln_b,
            const float* __restrict__ vw1, const float* __restrict__ vb1,
            const float* __restrict__ vw2, const float* __restrict__ vb2,
            const float* __restrict__ vw3, const float* __restrict__ vb3,
            float* __restrict__ out) {
    const int g = blockIdx.x, t = threadIdx.x;
    __shared__ float sx[256];
    __shared__ float sy[256];
    __shared__ float sr[256];

    const float x = pooled[(size_t)g * Hdim + t];

    sr[t] = x; __syncthreads();
    for (int s = 128; s > 0; s >>= 1) { if (t < s) sr[t] += sr[t + s]; __syncthreads(); }
    const float mu = sr[0] * (1.0f / 256.0f);
    __syncthreads();
    sr[t] = x * x; __syncthreads();
    for (int s = 128; s > 0; s >>= 1) { if (t < s) sr[t] += sr[t + s]; __syncthreads(); }
    const float var = sr[0] * (1.0f / 256.0f) - mu * mu;
    __syncthreads();

    sx[t] = (x - mu) * rsqrtf(var + 1e-5f) * ln_g[t] + ln_b[t];
    __syncthreads();

    float a1 = vb1[t];
    for (int k = 0; k < 256; ++k) a1 += sx[k] * vw1[k * 256 + t];
    sy[t] = gelu_exact(a1);
    __syncthreads();

    float y2 = 0.f;
    if (t < 128) {
        float a2 = vb2[t];
        for (int k = 0; k < 256; ++k) a2 += sy[k] * vw2[k * 128 + t];
        y2 = gelu_exact(a2) * vw3[t];
    }
    sr[t] = y2; __syncthreads();
    for (int s = 128; s > 0; s >>= 1) { if (t < s) sr[t] += sr[t + s]; __syncthreads(); }
    if (t == 0) out[g] = tanhf(sr[0] + vb3[0]);
}

// ============================================================================
extern "C" void kernel_launch(void* const* d_in, const int* in_sizes, int n_in,
                              void* d_out, int out_size, void* d_ws, size_t ws_size,
                              hipStream_t stream) {
    const float* nodef = (const float*)d_in[0];
    const float* edgef = (const float*)d_in[1];
    const int*   eidx  = (const int*)d_in[2];
    const int*   batch = (const int*)d_in[3];
    /* d_in[4] num_nodes_per_graph unused (constant 64) */
    const float* aw1 = (const float*)d_in[5];
    const float* ab1 = (const float*)d_in[6];
    const float* aw2 = (const float*)d_in[7];
    const float* ab2 = (const float*)d_in[8];
    const float* lng = (const float*)d_in[9];
    const float* lnb = (const float*)d_in[10];
    const float* vw1 = (const float*)d_in[11];
    const float* vb1 = (const float*)d_in[12];
    const float* vw2 = (const float*)d_in[13];
    const float* vb2 = (const float*)d_in[14];
    const float* vw3 = (const float*)d_in[15];
    const float* vb3 = (const float*)d_in[16];

    char* ws = (char*)d_ws;
    _Float16*     aw1h   = (_Float16*)ws;                              // 64 KB
    float*        scores = (float*)(ws + 65536);                       // ROWS*4
    int*          segarr = (int*)(ws + 65536 + (size_t)ROWS * 4);      // ROWS*4
    unsigned int* mcode  = (unsigned int*)(ws + 65536 + (size_t)ROWS * 8);
    float*        pooled = (float*)(ws + 65536 + (size_t)ROWS * 8 + 1024);

    prep_kernel<<<1, 256, 0, stream>>>(aw1, aw1h, mcode);
    attn_kernel<<<ROWS / 128, 256, 0, stream>>>(nodef, edgef, eidx, batch,
                                                aw1h, ab1, aw2, ab2,
                                                scores, segarr, mcode);
    pool_kernel<<<Ghm, 256, 0, stream>>>(nodef, edgef, scores, segarr, mcode, pooled);
    head_kernel<<<Ghm, 256, 0, stream>>>(pooled, lng, lnb, vw1, vb1, vw2, vb2,
                                         vw3, vb3, (float*)d_out);
}
